// ScaledDotProductAttention_40888088658342
// MI455X (gfx1250) — matrix operand order repaired and checked by reading, not compiled, not run on hardware
//
#include <hip/hip_runtime.h>
#include <stdint.h>

// ---------------------------------------------------------------------------
// Cosine self-attention, B=16, N=2048, D=512 (f32 in/out), MI455X gfx1250.
//   S = (x/|x|)(x/|x|)^T ; P = softmax(S) ; O = P x
// Pipeline:
//   K1a norm      : rownorm -> yhat = bf16(x/|x|) row-major [B,N,D]
//   K1b transpose : vt = bf16(x)^T [B,D,N] via LDS 64x64 tiles (coalesced both ways;
//                   avoids 2B-per-cacheline scatter ~64x write amplification)
//   K2  qk+softmax: S row-block (16 x 2048) via v_wmma_f32_16x16x32_bf16,
//                   softmax in registers, store P bf16 [B,N,N]
//   K3  pv        : O = P * V via WMMA, B-fragments contiguous from vt
// ~137 GFLOP of bf16 matmul (compute bound); P materialization adds ~256MB
// (~11us @ 23.3 TB/s) and is largely L2-resident (192MB L2).
// Workspace: yhat 32MB + vt 32MB + P 128MB = 192MB from d_ws.
// ---------------------------------------------------------------------------

#define BATCH 16
#define SEQ   2048
#define DIM   512

typedef __attribute__((ext_vector_type(16))) __bf16 v16bf;
typedef __attribute__((ext_vector_type(8)))  float  v8f;

union FragU { v16bf v; uint4 q[2]; };

__device__ __forceinline__ uint16_t f2bf(float f) {
  uint32_t u = __float_as_uint(f);
  u += 0x7FFFu + ((u >> 16) & 1u);      // round-to-nearest-even
  return (uint16_t)(u >> 16);
}

// ---------------------------------------------------------------------------
// K1a: per-row L2 norm -> yhat = bf16(x * rsqrt(sum x^2)). One wave per row.
// ---------------------------------------------------------------------------
__global__ __launch_bounds__(256)
void norm_kernel(const float* __restrict__ x, uint16_t* __restrict__ yhat) {
  const int wave = threadIdx.x >> 5;
  const int lane = threadIdx.x & 31;
  const int row  = blockIdx.x * 8 + wave;          // 0 .. B*N-1

  float f[16];
  const float4* xr = (const float4*)(x + (size_t)row * DIM + lane * 16);
  #pragma unroll
  for (int i = 0; i < 4; ++i) {
    float4 t = xr[i];
    f[4 * i + 0] = t.x; f[4 * i + 1] = t.y;
    f[4 * i + 2] = t.z; f[4 * i + 3] = t.w;
  }

  float ss = 0.f;
  #pragma unroll
  for (int i = 0; i < 16; ++i) ss += f[i] * f[i];
  #pragma unroll
  for (int m = 1; m < 32; m <<= 1) ss += __shfl_xor(ss, m, 32);
  const float rn = __builtin_amdgcn_rsqf(ss);      // 1/|x_row|

  uint32_t pk[8];
  #pragma unroll
  for (int i = 0; i < 8; ++i)
    pk[i] = (uint32_t)f2bf(f[2 * i] * rn) | ((uint32_t)f2bf(f[2 * i + 1] * rn) << 16);
  uint4* yd = (uint4*)(yhat + (size_t)row * DIM + lane * 16);
  yd[0] = make_uint4(pk[0], pk[1], pk[2], pk[3]);
  yd[1] = make_uint4(pk[4], pk[5], pk[6], pk[7]);
}

// ---------------------------------------------------------------------------
// K1b: vt[b][d][n] = bf16(x[b][n][d]) via LDS 64x64 tile.
// Loads coalesced along d (256B/row), stores coalesced along n (128B/row).
// ---------------------------------------------------------------------------
#define TT 64
__global__ __launch_bounds__(256)
void transpose_kernel(const float* __restrict__ x, uint16_t* __restrict__ vt) {
  __shared__ __align__(16) uint16_t tile[TT][TT + 8];   // +16B pad (stride 144B, 16B-mult)

  const int blk = blockIdx.x;
  const int b   = blk >> 8;                 // 8 d-tiles * 32 n-tiles = 256 per batch
  const int t2  = blk & 255;
  const int d0  = (t2 >> 5) * TT;
  const int n0  = (t2 & 31) * TT;
  const int tid = threadIdx.x;

  // load: 64 rows (n) x 64 cols (d) f32 -> bf16 transposed into LDS
  const int cg = tid & 15;                  // 16 column-groups of 4 floats
  const int rw = tid >> 4;                  // 16 rows per pass
  #pragma unroll
  for (int p = 0; p < 4; ++p) {
    const int lr = p * 16 + rw;             // local n
    float4 v = *(const float4*)(x + ((size_t)b * SEQ + n0 + lr) * DIM + d0 + cg * 4);
    tile[cg * 4 + 0][lr] = f2bf(v.x);
    tile[cg * 4 + 1][lr] = f2bf(v.y);
    tile[cg * 4 + 2][lr] = f2bf(v.z);
    tile[cg * 4 + 3][lr] = f2bf(v.w);
  }
  __syncthreads();

  // store: 64 d-rows x 64 n (128B per row), 32B per thread, fully coalesced
  const int dd  = tid >> 2;
  const int seg = tid & 3;
  uint16_t* dst = vt + ((size_t)b * DIM + d0 + dd) * SEQ + n0 + seg * 16;
  *(uint4*)(dst)     = *(const uint4*)&tile[dd][seg * 16];
  *(uint4*)(dst + 8) = *(const uint4*)&tile[dd][seg * 16 + 8];
}

// ---------------------------------------------------------------------------
// K2: one block = 16 query rows of one batch; 16 waves (512 thr). Wave w owns
// keys [128w, 128w+128) as 8 WMMA n-tiles; acc = 64 VGPRs -> better occupancy
// for hiding L2 latency of direct-global B fragments (batch tile is L2-hot:
// 2MB reused by 128 blocks). Softmax: shfl_xor in 16-lane halves + LDS reduce.
// ---------------------------------------------------------------------------
__global__ __launch_bounds__(512)
void qk_softmax_kernel(const uint16_t* __restrict__ yhat,
                       uint16_t* __restrict__ P) {
  const int wave = threadIdx.x >> 5;
  const int lane = threadIdx.x & 31;
  const int hf   = lane >> 4;                  // 16-lane half
  const int ln   = lane & 15;
  const int b    = blockIdx.x >> 7;            // / (SEQ/16)
  const int q0   = (blockIdx.x & 127) * 16;
  const size_t ybase = (size_t)b * SEQ * DIM;
  const int key0 = wave * 128;

  v8f acc[8];
  #pragma unroll
  for (int t = 0; t < 8; ++t)
    #pragma unroll
    for (int r = 0; r < 8; ++r) acc[t][r] = 0.f;

  // A fragment: query row m = ln; K-interleaving per ISA A-layout
  const uint16_t* arow = yhat + ybase + (size_t)(q0 + ln) * DIM;

  for (int kk = 0; kk < DIM; kk += 32) {
    __builtin_prefetch(arow + kk + 64, 0, 3);  // global_prefetch_b8, near-cache
    FragU a;
    a.q[0] = *(const uint4*)(arow + kk + hf * 8);        // K = half*8 .. +7
    a.q[1] = *(const uint4*)(arow + kk + 16 + hf * 8);   // K = 16+half*8 .. +7
    #pragma unroll
    for (int t = 0; t < 8; ++t) {
      // B(K=d, n=key) = yhat[key][d]: lane ln -> key, K = half*8..+7 and 16+half*8..+7
      const uint16_t* brow =
          yhat + ybase + (size_t)(key0 + t * 16 + ln) * DIM + kk + hf * 8;
      FragU bb;
      bb.q[0] = *(const uint4*)(brow);
      bb.q[1] = *(const uint4*)(brow + 16);
      acc[t] = __builtin_amdgcn_wmma_f32_16x16x32_bf16(
          false, a.v, false, bb.v, (short)0, acc[t], false, false);
    }
  }

  // ---- softmax over the full 2048-key row ----
  __shared__ float red[16][16];
  float rmax[8], rsum[8];

  #pragma unroll
  for (int r = 0; r < 8; ++r) {
    float m = acc[0][r];
    #pragma unroll
    for (int t = 1; t < 8; ++t) m = fmaxf(m, acc[t][r]);
    m = fmaxf(m, __shfl_xor(m, 1, 32));
    m = fmaxf(m, __shfl_xor(m, 2, 32));
    m = fmaxf(m, __shfl_xor(m, 4, 32));
    m = fmaxf(m, __shfl_xor(m, 8, 32));
    rmax[r] = m;                               // wave-local row max
  }
  if (ln == 0) {                               // lane0 -> rows 0-7, lane16 -> rows 8-15
    #pragma unroll
    for (int r = 0; r < 8; ++r) red[hf * 8 + r][wave] = rmax[r];
  }
  __syncthreads();
  #pragma unroll
  for (int r = 0; r < 8; ++r) {
    float m = red[hf * 8 + r][0];
    #pragma unroll
    for (int w2 = 1; w2 < 16; ++w2) m = fmaxf(m, red[hf * 8 + r][w2]);
    rmax[r] = m;                               // global row max
  }
  __syncthreads();

  const float LOG2E = 1.4426950408889634f;
  #pragma unroll
  for (int r = 0; r < 8; ++r) {
    float s = 0.f;
    #pragma unroll
    for (int t = 0; t < 8; ++t) {
      float e = __builtin_amdgcn_exp2f((acc[t][r] - rmax[r]) * LOG2E);
      acc[t][r] = e;
      s += e;
    }
    s += __shfl_xor(s, 1, 32);
    s += __shfl_xor(s, 2, 32);
    s += __shfl_xor(s, 4, 32);
    s += __shfl_xor(s, 8, 32);
    rsum[r] = s;
  }
  if (ln == 0) {
    #pragma unroll
    for (int r = 0; r < 8; ++r) red[hf * 8 + r][wave] = rsum[r];
  }
  __syncthreads();
  #pragma unroll
  for (int r = 0; r < 8; ++r) {
    float s = red[hf * 8 + r][0];
    #pragma unroll
    for (int w2 = 1; w2 < 16; ++w2) s += red[hf * 8 + r][w2];
    rsum[r] = 1.f / s;
  }

  // store P bf16: C/D layout (row m = 8*hf + r, col n = ln within tile t);
  // lanes are key-contiguous -> coalesces into 32B segments
  uint16_t* prow = P + ((size_t)b * SEQ + q0) * SEQ;
  #pragma unroll
  for (int t = 0; t < 8; ++t) {
    const int key = key0 + t * 16 + ln;
    #pragma unroll
    for (int r = 0; r < 8; ++r)
      prow[(size_t)(hf * 8 + r) * SEQ + key] = f2bf(acc[t][r] * rsum[r]);
  }
}

// ---------------------------------------------------------------------------
// K3: O = P * V. One block = 16 query rows x full D. 8 waves; wave w owns
// 64 output columns (4 n-tiles). B fragments read contiguously from vt.
// ---------------------------------------------------------------------------
__global__ __launch_bounds__(256)
void pv_kernel(const uint16_t* __restrict__ P,
               const uint16_t* __restrict__ vt,
               float* __restrict__ out) {
  const int wave = threadIdx.x >> 5;
  const int lane = threadIdx.x & 31;
  const int hf   = lane >> 4;
  const int ln   = lane & 15;
  const int b    = blockIdx.x >> 7;
  const int q0   = (blockIdx.x & 127) * 16;

  const uint16_t* arow = P + ((size_t)b * SEQ + q0 + ln) * SEQ;
  const size_t vtb = (size_t)b * DIM * SEQ;

  v8f acc[4];
  #pragma unroll
  for (int t = 0; t < 4; ++t)
    #pragma unroll
    for (int r = 0; r < 8; ++r) acc[t][r] = 0.f;

  for (int kk = 0; kk < SEQ; kk += 32) {
    __builtin_prefetch(arow + kk + 64, 0, 3);
    FragU a;
    a.q[0] = *(const uint4*)(arow + kk + hf * 8);
    a.q[1] = *(const uint4*)(arow + kk + 16 + hf * 8);
    #pragma unroll
    for (int t = 0; t < 4; ++t) {
      const int col = wave * 64 + t * 16 + ln;
      const uint16_t* brow = vt + vtb + (size_t)col * SEQ + kk + hf * 8;
      FragU bb;
      bb.q[0] = *(const uint4*)(brow);
      bb.q[1] = *(const uint4*)(brow + 16);
      acc[t] = __builtin_amdgcn_wmma_f32_16x16x32_bf16(
          false, a.v, false, bb.v, (short)0, acc[t], false, false);
    }
  }

  float* orow = out + ((size_t)b * SEQ + q0) * DIM;
  #pragma unroll
  for (int t = 0; t < 4; ++t)
    #pragma unroll
    for (int r = 0; r < 8; ++r)
      orow[(size_t)(hf * 8 + r) * DIM + wave * 64 + t * 16 + ln] = acc[t][r];
}

// ---------------------------------------------------------------------------
extern "C" void kernel_launch(void* const* d_in, const int* in_sizes, int n_in,
                              void* d_out, int out_size, void* d_ws, size_t ws_size,
                              hipStream_t stream) {
  (void)in_sizes; (void)n_in; (void)out_size; (void)ws_size;
  const float* x = (const float*)d_in[0];
  float* out = (float*)d_out;

  uint16_t* yhat = (uint16_t*)d_ws;                                  // 32 MB
  uint16_t* vt   = yhat + (size_t)BATCH * SEQ * DIM;                 // 32 MB
  uint16_t* P    = vt   + (size_t)BATCH * SEQ * DIM;                 // 128 MB

  norm_kernel<<<(BATCH * SEQ) / 8, 256, 0, stream>>>(x, yhat);
  transpose_kernel<<<BATCH * 256, 256, 0, stream>>>(x, vt);
  qk_softmax_kernel<<<BATCH * (SEQ / 16), 512, 0, stream>>>(yhat, P);
  pv_kernel<<<BATCH * (SEQ / 16), 256, 0, stream>>>(P, vt, out);
}
